// BahdanauAttention_2121713844647
// MI455X (gfx1250) — matrix operand order repaired and checked by reading, not compiled, not run on hardware
//
#include <hip/hip_runtime.h>
#include <math.h>

// ---------------- problem constants ----------------
constexpr int B_  = 64;
constexpr int S_  = 2048;
constexpr int D_  = 512;
constexpr int U_  = 256;

constexpr int CHUNK  = 64;          // sequence rows per score workgroup
constexpr int TILE_K = 32;          // K per WMMA step (bf16 16x16x32)
constexpr int VROW   = D_ + 8;      // padded bf16 row stride for values chunk (520)
constexpr int BROW   = TILE_K + 8;  // padded bf16 row stride for W1t panel (40)
constexpr int NKS    = D_ / TILE_K; // 16 K-steps

typedef __attribute__((ext_vector_type(16))) __bf16 v16bf;
typedef __attribute__((ext_vector_type(8)))  float  v8f;

union Frag { v16bf v; uint4 u[2]; };

__device__ __forceinline__ unsigned short f2bfbits(float x) {
    unsigned u = __float_as_uint(x);
    unsigned r = u + 0x7FFFu + ((u >> 16) & 1u);   // round-to-nearest-even
    return (unsigned short)(r >> 16);
}

__device__ __forceinline__ float fast_tanh(float x) {
#if defined(__has_builtin)
#if __has_builtin(__builtin_amdgcn_tanhf)
    return __builtin_amdgcn_tanhf(x);              // v_tanh_f32 (gfx1250 TRANS op)
#else
    return tanhf(x);
#endif
#else
    return tanhf(x);
#endif
}

// async global -> LDS copy of 64 contiguous bytes (4 x b128), per-lane addresses.
// INST_OFFSET is added to BOTH the LDS and global addresses (ISA 08_async_tensor §4.4).
__device__ __forceinline__ void async_copy64(unsigned lds_addr, const void* gaddr) {
    asm volatile(
        "global_load_async_to_lds_b128 %0, %1, off\n\t"
        "global_load_async_to_lds_b128 %0, %1, off offset:16\n\t"
        "global_load_async_to_lds_b128 %0, %1, off offset:32\n\t"
        "global_load_async_to_lds_b128 %0, %1, off offset:48"
        :: "v"(lds_addr), "v"(gaddr) : "memory");
}

__device__ __forceinline__ void wait_async0() {
    asm volatile("s_wait_asynccnt 0x0" ::: "memory");
}

// ---------------- kernel 1a: W1 [D,U] fp32 -> W1t [U,D] bf16 ----------------
__global__ __launch_bounds__(256)
void w1t_kernel(const float* __restrict__ W1, unsigned short* __restrict__ W1t) {
    int id = blockIdx.x * 256 + threadIdx.x;       // 0 .. U_*D_-1
    int u  = id >> 9;                              // / D_
    int k  = id & (D_ - 1);
    W1t[id] = f2bfbits(W1[k * U_ + u]);
}

// ---------------- kernel 1b: proj_q = query@W2 + b2 + b1 ----------------
__global__ __launch_bounds__(256)
void projq_kernel(const float* __restrict__ query, const float* __restrict__ W2,
                  const float* __restrict__ b2, const float* __restrict__ b1,
                  float* __restrict__ projq) {
    __shared__ float q[D_];
    int b = blockIdx.x, t = threadIdx.x;
    q[t]       = query[b * D_ + t];
    q[t + 256] = query[b * D_ + t + 256];
    __syncthreads();
    float s = 0.f;
    #pragma unroll 8
    for (int k = 0; k < D_; ++k) s = fmaf(q[k], W2[k * U_ + t], s);
    projq[b * U_ + t] = s + b2[t] + b1[t];
}

// ---------------- kernel 2: WMMA scores ----------------
// grid (S/CHUNK, B), block 256 (8 waves). Wave w: row block (w>>1)*16,
// column half (w&1)*128 -> 8 tiles of 16x16, K-loop of 16 x wmma_16x16x32_bf16.
// W1t K-panels are double-buffered in LDS via GLOBAL_LOAD_ASYNC_TO_LDS_B128.
__global__ __launch_bounds__(256)
void score_kernel(const float* __restrict__ values,
                  const unsigned short* __restrict__ W1t,
                  const float* __restrict__ projq,
                  const float* __restrict__ Vw,
                  const float* __restrict__ bV,
                  float* __restrict__ scores) {
    __shared__ alignas(16) unsigned short sV[CHUNK * VROW];   // 66,560 B
    __shared__ alignas(16) unsigned short sB[2][U_ * BROW];   // 2 x 20,480 B
    __shared__ float sQ[U_];
    __shared__ float sW[U_];
    __shared__ float sPart[8][16];

    const int b  = blockIdx.y;
    const int s0 = blockIdx.x * CHUNK;
    const int t  = threadIdx.x;
    const int lane = t & 31, wid = t >> 5;

    // issue async DMA of K-panel 0 first so it overlaps the fp32->bf16 conversion
    const unsigned sB0_addr = (unsigned)(unsigned long long)(const void*)&sB[0][t * BROW];
    const unsigned sB1_addr = (unsigned)(unsigned long long)(const void*)&sB[1][t * BROW];
    async_copy64(sB0_addr, W1t + (size_t)t * D_);

    sQ[t] = projq[b * U_ + t];
    sW[t] = Vw[t];

    // values[b, s0:s0+64, :] fp32 -> bf16 LDS (padded rows)
    const float4* src = (const float4*)(values + ((size_t)b * S_ + s0) * D_);
    for (int i = t; i < CHUNK * (D_ / 4); i += 256) {
        int r = i >> 7, c4 = i & 127;              // 128 float4 per row
        float4 f = src[i];
        ushort4 h;
        h.x = f2bfbits(f.x); h.y = f2bfbits(f.y);
        h.z = f2bfbits(f.z); h.w = f2bfbits(f.w);
        *(ushort4*)&sV[r * VROW + c4 * 4] = h;
    }
    wait_async0();
    __syncthreads();

    const int rb = wid >> 1;            // row block 0..3
    const int ch = wid & 1;             // column half 0..1
    const int arow  = rb * 16 + (lane & 15);
    const int ahalf = lane >> 4;

    v8f acc[8] = {};

    for (int kk = 0; kk < NKS; ++kk) {
        const int cur = kk & 1;
        if (kk + 1 < NKS) {
            // async DMA next W1t K-panel into the other buffer; overlaps the WMMAs
            async_copy64(cur ? sB0_addr : sB1_addr,
                         W1t + (size_t)t * D_ + (kk + 1) * TILE_K);
        }
        // A fragment: 16-bit 16x32 layout (two 16B runs per lane)
        Frag fa;
        fa.u[0] = *(const uint4*)&sV[arow * VROW + kk * TILE_K + ahalf * 8];
        fa.u[1] = *(const uint4*)&sV[arow * VROW + kk * TILE_K + 16 + ahalf * 8];
        #pragma unroll
        for (int j = 0; j < 8; ++j) {
            const int n0 = (ch * 8 + j) * 16;
            // B fragment: 16-bit 32x16 layout, lane n = lane&15, K runs 8*(lane>>4) and 16+8*(lane>>4)
            Frag fb;
            const uint4* bp = (const uint4*)&sB[cur][(n0 + (lane & 15)) * BROW + ahalf * 8];
            fb.u[0] = bp[0]; fb.u[1] = bp[2];
            acc[j] = __builtin_amdgcn_wmma_f32_16x16x32_bf16(
                         false, fa.v, false, fb.v, (short)0, acc[j], false, false);
        }
        wait_async0();      // own wave's DMA done
        __syncthreads();    // panel visible to all waves
    }

    // epilogue: rs[v] = sum_n tanh(acc + q[n]) * V[n] over this wave's 128 cols
    float rs[8] = {};
    #pragma unroll
    for (int j = 0; j < 8; ++j) {
        const int n = (ch * 8 + j) * 16 + (lane & 15);
        const float qn = sQ[n], wn = sW[n];
        #pragma unroll
        for (int v = 0; v < 8; ++v)
            rs[v] += fast_tanh(acc[j][v] + qn) * wn;
    }
    #pragma unroll
    for (int v = 0; v < 8; ++v) {
        #pragma unroll
        for (int m = 8; m >= 1; m >>= 1)
            rs[v] += __shfl_xor(rs[v], m, 32);
    }
    if ((lane & 15) == 0) {
        const int half = lane >> 4;
        #pragma unroll
        for (int v = 0; v < 8; ++v)
            sPart[wid][v + 8 * half] = rs[v];   // row v (+8) of block rb, wave's 128-col sum
    }
    __syncthreads();
    if (t < CHUNK) {
        const int rb2 = t >> 4, lr = t & 15;
        float sc = sPart[rb2 * 2][lr] + sPart[rb2 * 2 + 1][lr] + bV[0];
        scores[(size_t)b * S_ + s0 + t] = sc;
    }
}

// ---------------- kernel 3: softmax over S per batch ----------------
__global__ __launch_bounds__(256)
void softmax_kernel(const float* __restrict__ scores, float* __restrict__ attn) {
    __shared__ float red[256];
    const int b = blockIdx.x, t = threadIdx.x;
    const float* s = scores + (size_t)b * S_;
    float l[8]; float mx = -INFINITY;
    #pragma unroll
    for (int i = 0; i < 8; ++i) { l[i] = s[t + i * 256]; mx = fmaxf(mx, l[i]); }
    red[t] = mx; __syncthreads();
    for (int st = 128; st > 0; st >>= 1) { if (t < st) red[t] = fmaxf(red[t], red[t + st]); __syncthreads(); }
    mx = red[0]; __syncthreads();
    float sum = 0.f;
    #pragma unroll
    for (int i = 0; i < 8; ++i) { l[i] = __expf(l[i] - mx); sum += l[i]; }
    red[t] = sum; __syncthreads();
    for (int st = 128; st > 0; st >>= 1) { if (t < st) red[t] += red[t + st]; __syncthreads(); }
    const float inv = 1.f / red[0];
    float* a = attn + (size_t)b * S_;
    #pragma unroll
    for (int i = 0; i < 8; ++i) a[t + i * 256] = l[i] * inv;
}

// ---------------- kernel 4: context partials ----------------
// grid (16, B): chunk c covers 128 rows; thread t owns d = 2t, 2t+1 (coalesced float2)
__global__ __launch_bounds__(256)
void context_partial_kernel(const float* __restrict__ values,
                            const float* __restrict__ attn,
                            float* __restrict__ part) {
    __shared__ float sa[128];
    const int b = blockIdx.y, c = blockIdx.x, t = threadIdx.x;
    const int s0 = c * 128;
    if (t < 128) sa[t] = attn[(size_t)b * S_ + s0 + t];
    __syncthreads();
    const float2* vp = (const float2*)(values + ((size_t)b * S_ + s0) * D_) + t;
    float a0 = 0.f, a1 = 0.f;
    #pragma unroll 4
    for (int r = 0; r < 128; ++r) {
        float2 v = vp[(size_t)r * (D_ / 2)];
        a0 = fmaf(sa[r], v.x, a0);
        a1 = fmaf(sa[r], v.y, a1);
    }
    float* p = part + ((size_t)b * 16 + c) * D_ + t * 2;
    p[0] = a0; p[1] = a1;
}

// ---------------- kernel 5: reduce 16 partials -> context ----------------
__global__ __launch_bounds__(256)
void context_reduce_kernel(const float* __restrict__ part, float* __restrict__ out) {
    const int b = blockIdx.x, t = threadIdx.x;
    #pragma unroll
    for (int i = 0; i < 2; ++i) {
        const int d = t + i * 256;
        float s = 0.f;
        #pragma unroll
        for (int c = 0; c < 16; ++c) s += part[((size_t)b * 16 + c) * D_ + d];
        out[(size_t)b * D_ + d] = s;
    }
}

// ---------------- launch ----------------
extern "C" void kernel_launch(void* const* d_in, const int* in_sizes, int n_in,
                              void* d_out, int out_size, void* d_ws, size_t ws_size,
                              hipStream_t stream) {
    const float* query  = (const float*)d_in[0];
    const float* values = (const float*)d_in[1];
    const float* W1     = (const float*)d_in[2];
    const float* b1     = (const float*)d_in[3];
    const float* W2     = (const float*)d_in[4];
    const float* b2     = (const float*)d_in[5];
    const float* Vw     = (const float*)d_in[6];
    const float* bV     = (const float*)d_in[7];
    float* out = (float*)d_out;

    char* ws = (char*)d_ws;
    unsigned short* W1t = (unsigned short*)ws;                    // 256 KB
    float* projq  = (float*)(ws + (256u << 10));                  // 64 KB
    float* scores = (float*)(ws + (320u << 10));                  // 512 KB
    float* attn   = (float*)(ws + (832u << 10));                  // 512 KB
    float* part   = (float*)(ws + (1344u << 10));                 // 2 MB

    w1t_kernel<<<(U_ * D_) / 256, 256, 0, stream>>>(W1, W1t);
    projq_kernel<<<B_, 256, 0, stream>>>(query, W2, b2, b1, projq);
    score_kernel<<<dim3(S_ / CHUNK, B_), 256, 0, stream>>>(values, W1t, projq, Vw, bV, scores);
    softmax_kernel<<<B_, 256, 0, stream>>>(scores, attn);
    context_partial_kernel<<<dim3(16, B_), 256, 0, stream>>>(values, attn, part);
    context_reduce_kernel<<<B_, 256, 0, stream>>>(part, out);
}
